// LukeEntityAwareAttentionModel_9277129359591
// MI455X (gfx1250) — matrix operand order repaired and checked by reading, not compiled, not run on hardware
//
#include <hip/hip_runtime.h>
#include <math.h>

#define B_  16
#define S_  512
#define E_  128
#define D_  768
#define H_  12
#define DH_ 64
#define FF_ 3072
#define T_  (S_ + E_)

typedef __bf16 bf16_t;
typedef __attribute__((ext_vector_type(16))) __bf16 v16bf;
typedef __attribute__((ext_vector_type(8)))  __bf16 v8bf;
typedef __attribute__((ext_vector_type(8)))  float  v8f;
typedef __attribute__((ext_vector_type(4)))  unsigned int v4u;
typedef __attribute__((ext_vector_type(8)))  int v8i;
typedef __attribute__((ext_vector_type(4)))  int v4i;

#if defined(__has_builtin)
#if __has_builtin(__builtin_amdgcn_tensor_load_to_lds)
#define HAS_TDM 1
#endif
#endif
#ifndef HAS_TDM
#define HAS_TDM 0
#endif

__device__ __forceinline__ bf16_t tobf(float f) {
  union { float f; unsigned u; } v; v.f = f;
  unsigned r = (v.u + 0x7FFFu + ((v.u >> 16) & 1u)) >> 16;  // RNE
  unsigned short h = (unsigned short)r;
  bf16_t b;
  __builtin_memcpy(&b, &h, 2);
  return b;
}

// ---- Tensor Data Mover: one-instruction 2D tile load global->LDS ----------
// Loads `rows` x `rowlen` bf16 tile (row pitch `pitch` halves) with LDS row
// padding: pad_interval code `padi` (2^padi * 8B between pads), pad_amount
// code `pada` ((pada+1)*4B inserted). D# built per CDNA5 ISA 8.3/8.4.
// This toolchain's builtin is the 6-arg form:
//   (uint32x4 g0, int32x8 g1, int32x4 g2, int32x4 g3, int32x8 pad, i32 cpol)
__device__ __forceinline__ void tdm_load_tile(const bf16_t* g, unsigned lds_off,
                                              unsigned rows, unsigned rowlen,
                                              unsigned pitch, unsigned padi,
                                              unsigned pada)
{
#if HAS_TDM
  unsigned long long ga = (unsigned long long)(uintptr_t)g;
  v4u g0;
  g0[0] = 1u;                                   // count=1 (valid user D#)
  g0[1] = lds_off;                              // lds_addr (bytes)
  g0[2] = (unsigned)(ga & 0xFFFFFFFFu);         // global_addr lo
  g0[3] = (unsigned)((ga >> 32) & 0x01FFFFFFu) | (2u << 30);  // addr hi | type=2
  v8i g1;
  g1[0] = (int)((1u << 16) | (1u << 20) | (padi << 22) | (pada << 25)); // 2B, pad
  g1[1] = (int)((rowlen & 0xFFFFu) << 16);      // tensor_dim0 lo16
  g1[2] = (int)((rows   & 0xFFFFu) << 16);      // tensor_dim0 hi=0 | tensor_dim1 lo
  g1[3] = (int)((rowlen & 0xFFFFu) << 16);      // tensor_dim1 hi=0 | tile_dim0
  g1[4] = (int)(rows & 0xFFFFu);                // tile_dim1 | tile_dim2=0
  g1[5] = (int)pitch;                           // tensor_dim0_stride lo32
  g1[6] = 0;                                    // stride0 hi | stride1 lo (2D)
  g1[7] = 0;
  v4i z4 = {0, 0, 0, 0};
  v8i z8 = {0, 0, 0, 0, 0, 0, 0, 0};
  __builtin_amdgcn_tensor_load_to_lds(g0, g1, z4, z4, z8, 0);
#else
  (void)g; (void)lds_off; (void)rows; (void)rowlen; (void)pitch; (void)padi; (void)pada;
#endif
}

__device__ __forceinline__ void tensor_wait0() {
#if HAS_TDM
#if __has_builtin(__builtin_amdgcn_s_wait_tensorcnt)
  __builtin_amdgcn_s_wait_tensorcnt(0);
#else
  asm volatile("s_wait_tensorcnt 0x0" ::: "memory");
#endif
#endif
}

// A-fragment (16x32 bf16) from LDS row-major [row][k]; ISA 16-bit A layout.
__device__ __forceinline__ v16bf load_afrag(const bf16_t* base, int row, int stride,
                                            int kofs, int hi) {
  const bf16_t* p = base + row * stride + kofs + hi * 8;
  v8bf a0 = *(const v8bf*)p;
  v8bf a1 = *(const v8bf*)(p + 16);
  return __builtin_shufflevector(a0, a1, 0,1,2,3,4,5,6,7,8,9,10,11,12,13,14,15);
}
// B-fragment (32x16, KxN) from LDS stored [n][k]; lane holds col, ISA 16-bit layout (k=8h+s, 16+8h+s).
__device__ __forceinline__ v16bf load_bfrag(const bf16_t* base, int col, int stride,
                                            int kofs, int hi) {
  const bf16_t* p = base + col * stride + kofs + hi * 8;
  v8bf b0 = *(const v8bf*)p;
  v8bf b1 = *(const v8bf*)(p + 16);
  return __builtin_shufflevector(b0, b1, 0,1,2,3,4,5,6,7,8,9,10,11,12,13,14,15);
}

// ---------------------------------------------------------------------------
// GEMM: C[M,N] = act( A[M,K] @ Bt[N,K]^T + bias (+ R) ).  A,Bt bf16, C f32 or
// bf16. 256 thr = 8 waves, 128x128x32 tiles, TDM double-buffered staging.
// ---------------------------------------------------------------------------
#define BM 128
#define BN 128
#define BK 32
#define SA 40
#define SB 40

template<bool GELU, bool RES, bool OUTBF>
__global__ __launch_bounds__(256)
void gemm_kernel(const bf16_t* __restrict__ A, const bf16_t* __restrict__ Bt,
                 const float* __restrict__ bias, const float* __restrict__ Rres,
                 void* __restrict__ Cout, int M, int N, int K)
{
  __shared__ __align__(16) bf16_t lA[2][BM * SA];
  __shared__ __align__(16) bf16_t lB[2][BN * SB];

  const int t    = threadIdx.x;
  const int lane = t & 31;
  const int wid  = t >> 5;
  const int wm   = (wid >> 2) * 64;
  const int wn   = (wid & 3) * 32;
  const int hi   = lane >> 4;
  const int lo   = lane & 15;
  const int m0   = blockIdx.y * BM;
  const int n0   = blockIdx.x * BN;

  auto stage = [&](int k0, int buf) {
#if HAS_TDM
    if (wid == 0) {
      tdm_load_tile(A  + (size_t)m0 * K + k0, (unsigned)(uintptr_t)&lA[buf][0],
                    BM, BK, (unsigned)K, 3u, 3u);   // pad 64B rows -> stride 40h
      tdm_load_tile(Bt + (size_t)n0 * K + k0, (unsigned)(uintptr_t)&lB[buf][0],
                    BN, BK, (unsigned)K, 3u, 3u);
    }
#else
    #pragma unroll
    for (int i = 0; i < 2; ++i) {
      int idx = t + 256 * i;
      int r = idx >> 2, c = (idx & 3) * 8;
      *(v8bf*)&lA[buf][r * SA + c] = *(const v8bf*)&A [(size_t)(m0 + r) * K + k0 + c];
      *(v8bf*)&lB[buf][r * SB + c] = *(const v8bf*)&Bt[(size_t)(n0 + r) * K + k0 + c];
    }
#endif
  };

  v8f acc[4][2] = {};
  stage(0, 0);
  const int nsteps = K / BK;
  for (int s = 0; s < nsteps; ++s) {
    const int buf = s & 1;
#if HAS_TDM
    if (wid == 0) tensor_wait0();
#endif
    __syncthreads();
    if (s + 1 < nsteps) stage((s + 1) * BK, buf ^ 1);

    const bf16_t* cA = &lA[buf][0];
    const bf16_t* cB = &lB[buf][0];
    v16bf af[4];
    #pragma unroll
    for (int mi = 0; mi < 4; ++mi)
      af[mi] = load_afrag(cA, wm + mi * 16 + lo, SA, 0, hi);
    #pragma unroll
    for (int ni = 0; ni < 2; ++ni) {
      v16bf bfv = load_bfrag(cB, wn + ni * 16 + lo, SB, 0, hi);
      #pragma unroll
      for (int mi = 0; mi < 4; ++mi)
        acc[mi][ni] = __builtin_amdgcn_wmma_f32_16x16x32_bf16(
            false, af[mi], false, bfv, (short)0, acc[mi][ni], false, false);
    }
    __syncthreads();
  }

  #pragma unroll
  for (int mi = 0; mi < 4; ++mi)
    #pragma unroll
    for (int ni = 0; ni < 2; ++ni)
      #pragma unroll
      for (int v = 0; v < 8; ++v) {
        int row = m0 + wm + mi * 16 + v + 8 * hi;
        int col = n0 + wn + ni * 16 + lo;
        float x = acc[mi][ni][v] + bias[col];
        if (RES)  x += Rres[(size_t)row * N + col];
        if (GELU) x = 0.5f * x * (1.0f + erff(x * 0.70710678118654752f));
        if (OUTBF) ((bf16_t*)Cout)[(size_t)row * N + col] = tobf(x);
        else       ((float*) Cout)[(size_t)row * N + col] = x;
      }
}

// ---------------------------------------------------------------------------
// Entity-aware flash attention (bf16 in, bf16 ctx out). 128 thr = 4 waves,
// 64 q/block, 64-key tiles, Q/K tiles staged by TDM, V transposed manually.
// ---------------------------------------------------------------------------
__global__ __launch_bounds__(128)
void attn_kernel(const bf16_t* __restrict__ qww, const bf16_t* __restrict__ qwe,
                 const bf16_t* __restrict__ qew, const bf16_t* __restrict__ qee,
                 const bf16_t* __restrict__ kall, const bf16_t* __restrict__ vall,
                 const float* __restrict__ mask, bf16_t* __restrict__ ctx)
{
  __shared__ __align__(16) bf16_t sQw[64 * 72];
  __shared__ __align__(16) bf16_t sQe[64 * 72];
  __shared__ __align__(16) bf16_t sK [64 * 72];   // [key][d]
  __shared__ __align__(16) bf16_t sV [64 * 72];   // [d][key]
  __shared__ __align__(16) bf16_t sP [64 * 72];   // [q][key]

  const int t    = threadIdx.x;
  const int lane = t & 31;
  const int wave = t >> 5;
  const int hi   = lane >> 4, lo = lane & 15;
  const int bh   = blockIdx.y;
  const int b    = bh / H_, h = bh % H_;
  const int qbase = blockIdx.x * 64;

  const bool entq = (qbase >= S_);
  const bf16_t* Qa = entq ? qew : qww;
  const bf16_t* Qb = entq ? qee : qwe;
  const int qrow0 = entq ? (b * E_ + (qbase - S_)) : (b * S_ + qbase);

#if HAS_TDM
  if (wave == 0) {
    tdm_load_tile(Qa + (size_t)qrow0 * D_ + h * DH_, (unsigned)(uintptr_t)sQw,
                  64, 64, D_, 4u, 3u);            // pad 128B rows -> stride 72h
    tdm_load_tile(Qb + (size_t)qrow0 * D_ + h * DH_, (unsigned)(uintptr_t)sQe,
                  64, 64, D_, 4u, 3u);
  }
#else
  #pragma unroll
  for (int i = 0; i < 4; ++i) {
    int idx = t + 128 * i;
    int r = idx >> 3, dc = (idx & 7) * 8;
    size_t off = (size_t)(qrow0 + r) * D_ + h * DH_ + dc;
    *(v8bf*)&sQw[r * 72 + dc] = *(const v8bf*)&Qa[off];
    *(v8bf*)&sQe[r * 72 + dc] = *(const v8bf*)&Qb[off];
  }
#endif

  float m_r[8], l_r[8];
  v8f oacc[4] = {};
  #pragma unroll
  for (int v = 0; v < 8; ++v) { m_r[v] = -1e30f; l_r[v] = 0.0f; }

  for (int kt = 0; kt < T_ / 64; ++kt) {
    const int kbase = kt * 64;
    __syncthreads();
#if HAS_TDM
    if (wave == 0)
      tdm_load_tile(kall + (size_t)(b * T_ + kbase) * D_ + h * DH_,
                    (unsigned)(uintptr_t)sK, 64, 64, D_, 4u, 3u);
#else
    #pragma unroll
    for (int i = 0; i < 4; ++i) {
      int idx = t + 128 * i;
      int r = idx >> 3, dc = (idx & 7) * 8;
      *(v8bf*)&sK[r * 72 + dc] =
          *(const v8bf*)&kall[(size_t)(b * T_ + kbase + r) * D_ + h * DH_ + dc];
    }
#endif
    // V tile transposed [d][key] (2-byte granular -> manual scatter)
    #pragma unroll
    for (int i = 0; i < 4; ++i) {
      int idx = t + 128 * i;
      int key = idx >> 3, dc = (idx & 7) * 8;
      v8bf vv = *(const v8bf*)&vall[(size_t)(b * T_ + kbase + key) * D_ + h * DH_ + dc];
      #pragma unroll
      for (int e = 0; e < 8; ++e) sV[(dc + e) * 72 + key] = vv[e];
    }
#if HAS_TDM
    if (wave == 0) tensor_wait0();
#endif
    __syncthreads();

    const bf16_t* sQ = (kbase < S_) ? sQw : sQe;  // entity-aware Q select

    v8f sacc[4] = {};
    #pragma unroll
    for (int ks = 0; ks < 64; ks += 32) {
      v16bf af = load_afrag(sQ, wave * 16 + lo, 72, ks, hi);
      #pragma unroll
      for (int nt = 0; nt < 4; ++nt) {
        v16bf bfv = load_bfrag(sK, nt * 16 + lo, 72, ks, hi);
        sacc[nt] = __builtin_amdgcn_wmma_f32_16x16x32_bf16(
            false, af, false, bfv, (short)0, sacc[nt], false, false);
      }
    }

    float sv[4][8];
    #pragma unroll
    for (int nt = 0; nt < 4; ++nt) {
      float mk = mask[(size_t)b * T_ + kbase + nt * 16 + lo];
      #pragma unroll
      for (int v = 0; v < 8; ++v) sv[nt][v] = sacc[nt][v] * 0.125f + mk;
    }

    #pragma unroll
    for (int v = 0; v < 8; ++v) {
      float rm = fmaxf(fmaxf(sv[0][v], sv[1][v]), fmaxf(sv[2][v], sv[3][v]));
      #pragma unroll
      for (int off = 1; off < 16; off <<= 1) rm = fmaxf(rm, __shfl_xor(rm, off, 32));
      float mn = fmaxf(m_r[v], rm);
      float alpha = __expf(m_r[v] - mn);
      m_r[v] = mn;
      float rs = 0.0f;
      #pragma unroll
      for (int nt = 0; nt < 4; ++nt) {
        float p = __expf(sv[nt][v] - mn);
        sv[nt][v] = p;
        rs += p;
      }
      #pragma unroll
      for (int off = 1; off < 16; off <<= 1) rs += __shfl_xor(rs, off, 32);
      l_r[v] = l_r[v] * alpha + rs;
      #pragma unroll
      for (int nt = 0; nt < 4; ++nt) oacc[nt][v] *= alpha;
      #pragma unroll
      for (int nt = 0; nt < 4; ++nt)
        sP[(wave * 16 + v + 8 * hi) * 72 + nt * 16 + lo] = tobf(sv[nt][v]);
    }
    asm volatile("s_wait_dscnt 0" ::: "memory");

    #pragma unroll
    for (int ks = 0; ks < 64; ks += 32) {
      v16bf af = load_afrag(sP, wave * 16 + lo, 72, ks, hi);
      #pragma unroll
      for (int nt = 0; nt < 4; ++nt) {
        v16bf bfv = load_bfrag(sV, nt * 16 + lo, 72, ks, hi);
        oacc[nt] = __builtin_amdgcn_wmma_f32_16x16x32_bf16(
            false, af, false, bfv, (short)0, oacc[nt], false, false);
      }
    }
  }

  #pragma unroll
  for (int nt = 0; nt < 4; ++nt)
    #pragma unroll
    for (int v = 0; v < 8; ++v) {
      int row = qbase + wave * 16 + v + 8 * hi;
      int col = nt * 16 + lo;
      ctx[(size_t)(b * T_ + row) * D_ + h * DH_ + col] = tobf(oacc[nt][v] / l_r[v]);
    }
}

// ---------------------------------------------------------------------------
// Row LayerNorm over D=768; optional bf16 shadow copy; optional output split.
// ---------------------------------------------------------------------------
__global__ __launch_bounds__(256)
void ln_kernel(const float* __restrict__ x, const float* __restrict__ g,
               const float* __restrict__ bta, float* __restrict__ outw,
               float* __restrict__ oute, bf16_t* __restrict__ bout, int split)
{
  __shared__ float red[256];
  const int row = blockIdx.x;
  const int t = threadIdx.x;
  const float* xr = x + (size_t)row * D_;
  float v0 = xr[t], v1 = xr[t + 256], v2 = xr[t + 512];
  red[t] = v0 + v1 + v2;
  __syncthreads();
  for (int o = 128; o > 0; o >>= 1) { if (t < o) red[t] += red[t + o]; __syncthreads(); }
  float mean = red[0] * (1.0f / D_);
  __syncthreads();
  float d0 = v0 - mean, d1 = v1 - mean, d2 = v2 - mean;
  red[t] = d0 * d0 + d1 * d1 + d2 * d2;
  __syncthreads();
  for (int o = 128; o > 0; o >>= 1) { if (t < o) red[t] += red[t + o]; __syncthreads(); }
  float inv = rsqrtf(red[0] * (1.0f / D_) + 1e-12f);

  float r0 = d0 * inv * g[t]       + bta[t];
  float r1 = d1 * inv * g[t + 256] + bta[t + 256];
  float r2 = d2 * inv * g[t + 512] + bta[t + 512];

  float* dst;
  if (split) {
    int bb = row / T_, tt = row % T_;
    dst = (tt < S_) ? (outw + (size_t)(bb * S_ + tt) * D_)
                    : (oute + (size_t)(bb * E_ + (tt - S_)) * D_);
  } else {
    dst = outw + (size_t)row * D_;
  }
  dst[t] = r0; dst[t + 256] = r1; dst[t + 512] = r2;
  if (bout) {
    bf16_t* bd = bout + (size_t)row * D_;
    bd[t] = tobf(r0); bd[t + 256] = tobf(r1); bd[t + 512] = tobf(r2);
  }
}

// concat word|entity -> kv (f32) and kvb (bf16)
__global__ void concat_kernel(const float* __restrict__ w, const float* __restrict__ e,
                              float* __restrict__ kv, bf16_t* __restrict__ kvb)
{
  size_t i = (size_t)blockIdx.x * blockDim.x + threadIdx.x;
  int d = (int)(i % D_);
  size_t td = i / D_;
  int tt = (int)(td % T_);
  int b  = (int)(td / T_);
  float v = (tt < S_) ? w[((size_t)b * S_ + tt) * D_ + d]
                      : e[((size_t)b * E_ + (tt - S_)) * D_ + d];
  kv[i] = v;
  kvb[i] = tobf(v);
}

// plain f32 -> bf16 convert (4 elems/thread)
__global__ void cvt_kernel(const float* __restrict__ in, bf16_t* __restrict__ out)
{
  size_t i = ((size_t)blockIdx.x * blockDim.x + threadIdx.x) * 4;
  #pragma unroll
  for (int e = 0; e < 4; ++e) out[i + e] = tobf(in[i + e]);
}

// transpose + convert: W[K][N] f32 -> Wt[N][K] bf16, 64x64 LDS tiles
__global__ __launch_bounds__(256)
void wtcvt_kernel(const float* __restrict__ in, bf16_t* __restrict__ out, int K, int N)
{
  __shared__ float tile[64 * 65];
  const int t = threadIdx.x;
  const int k0 = blockIdx.y * 64;
  const int n0 = blockIdx.x * 64;
  #pragma unroll
  for (int i = 0; i < 16; ++i) {
    int idx = t + 256 * i;
    int r = idx >> 6, c = idx & 63;
    tile[r * 65 + c] = in[(size_t)(k0 + r) * N + n0 + c];
  }
  __syncthreads();
  #pragma unroll
  for (int i = 0; i < 16; ++i) {
    int idx = t + 256 * i;
    int r = idx >> 6, c = idx & 63;          // r: n index, c: k index
    out[(size_t)(n0 + r) * K + k0 + c] = tobf(tile[c * 65 + r]);
  }
}

// ---------------------------------------------------------------------------
extern "C" void kernel_launch(void* const* d_in, const int* in_sizes, int n_in,
                              void* d_out, int out_size, void* d_ws, size_t ws_size,
                              hipStream_t stream)
{
  (void)in_sizes; (void)n_in; (void)out_size; (void)ws_size;
  const float* word = (const float*)d_in[0];
  const float* ent  = (const float*)d_in[1];
  const float* mask = (const float*)d_in[2];
  const float* W_q  = (const float*)d_in[3];   const float* b_q   = (const float*)d_in[4];
  const float* W_k  = (const float*)d_in[5];   const float* b_k   = (const float*)d_in[6];
  const float* W_v  = (const float*)d_in[7];   const float* b_v   = (const float*)d_in[8];
  const float* W_w2e= (const float*)d_in[9];   const float* b_w2e = (const float*)d_in[10];
  const float* W_e2w= (const float*)d_in[11];  const float* b_e2w = (const float*)d_in[12];
  const float* W_e2e= (const float*)d_in[13];  const float* b_e2e = (const float*)d_in[14];
  const float* W_ao = (const float*)d_in[15];  const float* b_ao  = (const float*)d_in[16];
  const float* g_ao = (const float*)d_in[17];  const float* be_ao = (const float*)d_in[18];
  const float* W_i  = (const float*)d_in[19];  const float* b_i   = (const float*)d_in[20];
  const float* W_o  = (const float*)d_in[21];  const float* b_o   = (const float*)d_in[22];
  const float* g_o  = (const float*)d_in[23];  const float* be_o  = (const float*)d_in[24];

  const size_t BTD = (size_t)B_ * T_ * D_;
  const size_t BSD = (size_t)B_ * S_ * D_;
  const size_t BED = (size_t)B_ * E_ * D_;
  const size_t DD  = (size_t)D_ * D_;
  const size_t DF  = (size_t)D_ * FF_;

  float*  kv     = (float*)d_ws;
  float*  aopre  = kv + BTD;          // reused as final pre-LN buffer
  float*  ao     = aopre + BTD;
  bf16_t* kvb    = (bf16_t*)(ao + BTD);
  bf16_t* aob    = kvb + BTD;
  bf16_t* wordb  = aob + BTD;
  bf16_t* entb   = wordb + BSD;
  bf16_t* ctxb   = entb + BED;
  bf16_t* wt_q   = ctxb + BTD;
  bf16_t* wt_k   = wt_q  + DD;
  bf16_t* wt_v   = wt_k  + DD;
  bf16_t* wt_w2e = wt_v  + DD;
  bf16_t* wt_e2w = wt_w2e+ DD;
  bf16_t* wt_e2e = wt_e2w+ DD;
  bf16_t* wt_ao  = wt_e2e+ DD;
  bf16_t* wt_i   = wt_ao + DD;
  bf16_t* wt_o   = wt_i  + DF;
  bf16_t* qwwb   = wt_o  + DF;
  bf16_t* qweb   = qwwb + BSD;
  bf16_t* qewb   = qweb + BSD;
  bf16_t* qeeb   = qewb + BED;
  bf16_t* kallb  = qeeb + BED;
  bf16_t* vallb  = kallb + BTD;
  bf16_t* interb = qwwb;              // overlay: 2B(S+E+T)D halves == B*T*FF

  dim3 blk(256);

  // weight transpose+convert (f32 [K][N] -> bf16 [N][K])
  wtcvt_kernel<<<dim3(D_/64,  D_/64),  blk, 0, stream>>>(W_q,   wt_q,   D_,  D_);
  wtcvt_kernel<<<dim3(D_/64,  D_/64),  blk, 0, stream>>>(W_k,   wt_k,   D_,  D_);
  wtcvt_kernel<<<dim3(D_/64,  D_/64),  blk, 0, stream>>>(W_v,   wt_v,   D_,  D_);
  wtcvt_kernel<<<dim3(D_/64,  D_/64),  blk, 0, stream>>>(W_w2e, wt_w2e, D_,  D_);
  wtcvt_kernel<<<dim3(D_/64,  D_/64),  blk, 0, stream>>>(W_e2w, wt_e2w, D_,  D_);
  wtcvt_kernel<<<dim3(D_/64,  D_/64),  blk, 0, stream>>>(W_e2e, wt_e2e, D_,  D_);
  wtcvt_kernel<<<dim3(D_/64,  D_/64),  blk, 0, stream>>>(W_ao,  wt_ao,  D_,  D_);
  wtcvt_kernel<<<dim3(FF_/64, D_/64),  blk, 0, stream>>>(W_i,   wt_i,   D_,  FF_);
  wtcvt_kernel<<<dim3(D_/64,  FF_/64), blk, 0, stream>>>(W_o,   wt_o,   FF_, D_);

  cvt_kernel<<<dim3((unsigned)(BSD / 1024)), blk, 0, stream>>>(word, wordb);
  cvt_kernel<<<dim3((unsigned)(BED / 1024)), blk, 0, stream>>>(ent,  entb);
  concat_kernel<<<dim3((unsigned)(BTD / 256)), blk, 0, stream>>>(word, ent, kv, kvb);

  auto gemm = [&](const bf16_t* A, const bf16_t* Bt, const float* bias, const float* R,
                  void* C, int M, int N, int K, int mode) {
    dim3 g(N / BN, M / BM);
    if (mode == 0)      gemm_kernel<false, false, true ><<<g, blk, 0, stream>>>(A, Bt, bias, R, C, M, N, K);
    else if (mode == 1) gemm_kernel<true,  false, true ><<<g, blk, 0, stream>>>(A, Bt, bias, R, C, M, N, K);
    else                gemm_kernel<false, true,  false><<<g, blk, 0, stream>>>(A, Bt, bias, R, C, M, N, K);
  };

  gemm(wordb, wt_q,   b_q,   nullptr, qwwb,  B_ * S_, D_, D_, 0);
  gemm(wordb, wt_w2e, b_w2e, nullptr, qweb,  B_ * S_, D_, D_, 0);
  gemm(entb,  wt_e2w, b_e2w, nullptr, qewb,  B_ * E_, D_, D_, 0);
  gemm(entb,  wt_e2e, b_e2e, nullptr, qeeb,  B_ * E_, D_, D_, 0);
  gemm(kvb,   wt_k,   b_k,   nullptr, kallb, B_ * T_, D_, D_, 0);
  gemm(kvb,   wt_v,   b_v,   nullptr, vallb, B_ * T_, D_, D_, 0);

  attn_kernel<<<dim3(T_ / 64, B_ * H_), dim3(128), 0, stream>>>(
      qwwb, qweb, qewb, qeeb, kallb, vallb, mask, ctxb);

  gemm(ctxb, wt_ao, b_ao, kv, aopre, B_ * T_, D_, D_, 2);
  ln_kernel<<<dim3(B_ * T_), blk, 0, stream>>>(aopre, g_ao, be_ao, ao, nullptr, aob, 0);
  gemm(aob, wt_i, b_i, nullptr, interb, B_ * T_, FF_, D_, 1);
  gemm(interb, wt_o, b_o, ao, aopre, B_ * T_, D_, FF_, 2);

  float* outw = (float*)d_out;
  float* oute = outw + BSD;
  ln_kernel<<<dim3(B_ * T_), blk, 0, stream>>>(aopre, g_o, be_o, outw, oute, nullptr, 1);
}
